// Encoder_Model_89369679495588
// MI455X (gfx1250) — matrix operand order repaired and checked by reading, not compiled, not run on hardware
//
#include <hip/hip_runtime.h>

// ---------------- problem constants (from reference) ----------------
#define BB 8
#define SS 1024
#define DD 512
#define HH 8
#define DK 64
#define FFD 2048
#define EPS 1e-5f
#define LN_BLOCKS 4096  // 4096 * 256 * 4 = 4,194,304 = B*S*D

typedef __attribute__((ext_vector_type(16))) _Float16 f16x16;
typedef __attribute__((ext_vector_type(8)))  _Float16 f16x8;
typedef __attribute__((ext_vector_type(4)))  _Float16 f16x4;
typedef __attribute__((ext_vector_type(8)))  float    f32x8;

__device__ inline f32x8 vzero8() {
  f32x8 z;
#pragma unroll
  for (int i = 0; i < 8; ++i) z[i] = 0.0f;
  return z;
}

// =====================================================================
// Elementwise f32 -> f16 (vectorized: b128 load, b64 store)
// =====================================================================
__global__ __launch_bounds__(256) void cvt_f16_kernel(
    const float* __restrict__ src, _Float16* __restrict__ dst) {
  size_t i0 = ((size_t)blockIdx.x * 256 + threadIdx.x) * 4;
  float4 v = *(const float4*)(src + i0);
  f16x4 h;
  h[0] = (_Float16)v.x; h[1] = (_Float16)v.y;
  h[2] = (_Float16)v.z; h[3] = (_Float16)v.w;
  *(f16x4*)(dst + i0) = h;
}

// =====================================================================
// Weight transpose + convert:  WT[n][k] (f16) = W[k][n] (f32)
// 32x32 tiles through LDS; both global sides coalesced.
// =====================================================================
__global__ __launch_bounds__(256) void wtrans_kernel(
    const float* __restrict__ W, _Float16* __restrict__ WT, int K, int N) {
  __shared__ float t[32][33];
  const int n0 = blockIdx.x * 32, k0 = blockIdx.y * 32;
  const int x = threadIdx.x & 31, y4 = threadIdx.x >> 5;
#pragma unroll
  for (int i = 0; i < 4; ++i) {
    int ky = y4 * 4 + i;
    t[ky][x] = W[(size_t)(k0 + ky) * N + n0 + x];
  }
  __syncthreads();
#pragma unroll
  for (int i = 0; i < 4; ++i) {
    int ny = y4 * 4 + i;
    WT[(size_t)(n0 + ny) * K + k0 + x] = (_Float16)t[x][ny];
  }
}

// =====================================================================
// f16 GEMM:  out[M,N] = AH[M,K] @ WT[N,K]^T + bias
// Block 256 thr = 8 waves (2M x 4N), each wave 32x32 (2x2 WMMA tiles).
// Block tile 64x128, K chunks of 32, register double buffering.
// Outputs: optional f32, optional f16 (row-major), optional f16
// head-transposed VT[b,h,d,s] (for attention V).
// =====================================================================
__global__ __launch_bounds__(256) void gemm_f16_kernel(
    const _Float16* __restrict__ AH, const _Float16* __restrict__ WT,
    const float* __restrict__ bias,
    float* __restrict__ outF, _Float16* __restrict__ outH,
    _Float16* __restrict__ outVT,
    int M, int N, int K, int relu) {
  __shared__ __attribute__((aligned(16))) _Float16 la[64 * 40];   // [m][k]
  __shared__ __attribute__((aligned(16))) _Float16 lb[128 * 40];  // [n][k]

  const int tid  = threadIdx.x;
  const int wave = tid >> 5, lane = tid & 31;
  const int g = lane >> 4, c = lane & 15;
  const int wm = wave >> 2, wn = wave & 3;
  const int m0 = blockIdx.y * 64, n0 = blockIdx.x * 128;

  // staging assignment: A 64 rows x 4 thr (8 halves each), B 128 rows x 2 thr
  const int am = tid >> 2, ak = (tid & 3) * 8;
  const int bn = tid >> 1, bk = (tid & 1) * 16;
  const _Float16* aptr = AH + (size_t)(m0 + am) * K + ak;
  const _Float16* bptr = WT + (size_t)(n0 + bn) * K + bk;

  // preload chunk 0 into registers
  f16x8 ra  = *(const f16x8*)(aptr);
  f16x8 rb0 = *(const f16x8*)(bptr);
  f16x8 rb1 = *(const f16x8*)(bptr + 8);

  f32x8 acc[2][2];
#pragma unroll
  for (int mt = 0; mt < 2; ++mt)
#pragma unroll
    for (int nt = 0; nt < 2; ++nt) acc[mt][nt] = vzero8();

  for (int k0 = 0; k0 < K; k0 += 32) {
    __syncthreads();  // WAR guard on LDS
    *(f16x8*)(la + am * 40 + ak) = ra;
    *(f16x8*)(lb + bn * 40 + bk) = rb0;
    *(f16x8*)(lb + bn * 40 + bk + 8) = rb1;
    __syncthreads();
    if (k0 + 32 < K) {  // overlap next-chunk loads with WMMA below
      ra  = *(const f16x8*)(aptr + k0 + 32);
      rb0 = *(const f16x8*)(bptr + k0 + 32);
      rb1 = *(const f16x8*)(bptr + k0 + 32 + 8);
    }
    f16x16 af[2], bf[2];
#pragma unroll
    for (int mt = 0; mt < 2; ++mt) {  // A frag: K run [16g..16g+15]
      const int r = wm * 32 + mt * 16 + c;
      const f16x8 lo = *(const f16x8*)(la + r * 40 + 16 * g);
      const f16x8 hi = *(const f16x8*)(la + r * 40 + 16 * g + 8);
#pragma unroll
      for (int e = 0; e < 8; ++e) { af[mt][e] = lo[e]; af[mt][e + 8] = hi[e]; }
    }
#pragma unroll
    for (int nt = 0; nt < 2; ++nt) {  // B frag: K run [16g..16g+15]
      const int n = wn * 32 + nt * 16 + c;
      const f16x8 lo = *(const f16x8*)(lb + n * 40 + 16 * g);
      const f16x8 hi = *(const f16x8*)(lb + n * 40 + 16 * g + 8);
#pragma unroll
      for (int e = 0; e < 8; ++e) { bf[nt][e] = lo[e]; bf[nt][e + 8] = hi[e]; }
    }
#pragma unroll
    for (int mt = 0; mt < 2; ++mt)
#pragma unroll
      for (int nt = 0; nt < 2; ++nt)
        acc[mt][nt] = __builtin_amdgcn_wmma_f32_16x16x32_f16(
            false, af[mt], false, bf[nt], (short)0, acc[mt][nt], false, false);
  }

  // epilogue
#pragma unroll
  for (int mt = 0; mt < 2; ++mt)
#pragma unroll
    for (int nt = 0; nt < 2; ++nt) {
      const int col = n0 + wn * 32 + nt * 16 + c;
      const float bb = bias[col];
#pragma unroll
      for (int i = 0; i < 8; ++i) {
        const int row = m0 + wm * 32 + mt * 16 + 8 * g + i;
        float v = acc[mt][nt][i] + bb;
        if (relu) v = fmaxf(v, 0.0f);
        if (outF) outF[(size_t)row * N + col] = v;
        if (outH) outH[(size_t)row * N + col] = (_Float16)v;
        if (outVT) {  // VT[((b*H + h)*DK + d)*S + s]
          const int bI = row >> 10, s = row & 1023;
          const int h = col >> 6, d = col & 63;
          outVT[((size_t)((bI * HH + h) * DK + d)) * SS + s] = (_Float16)v;
        }
      }
    }
}

// =====================================================================
// Flash attention, all-f16 inputs. 1 wave = 16 queries x head (d=64).
// Block = 4 waves per (b,h). Key chunks of 32. V comes pre-transposed
// (VT[b,h,d,s]) so every fragment is contiguous b128 loads.
// Scores scaled by 1/32 post-WMMA (reference scale = (D/H)/2).
// =====================================================================
__global__ __launch_bounds__(128) void flash_attn_kernel(
    const _Float16* __restrict__ qb, const _Float16* __restrict__ kb,
    const _Float16* __restrict__ vt, _Float16* __restrict__ ctx) {
  __shared__ __attribute__((aligned(16))) _Float16 pl[4 * 16 * 40];
  const int b = blockIdx.z, h = blockIdx.y;
  const int wave = threadIdx.x >> 5, lane = threadIdx.x & 31;
  const int g = lane >> 4, c = lane & 15;
  const int q0 = blockIdx.x * 64 + wave * 16;
  _Float16* pw = pl + wave * 16 * 40;

  // Q fragments (2 d-chunks of 32)
  const _Float16* qrow = qb + (size_t)(b * SS + q0 + c) * DD + h * DK;
  f16x16 aq[2];
#pragma unroll
  for (int dc = 0; dc < 2; ++dc) {
    const f16x8 lo = *(const f16x8*)(qrow + 32 * dc + 16 * g);
    const f16x8 hi = *(const f16x8*)(qrow + 32 * dc + 16 * g + 8);
#pragma unroll
    for (int e = 0; e < 8; ++e) { aq[dc][e] = lo[e]; aq[dc][e + 8] = hi[e]; }
  }

  float mrow[8], lrow[8];
  f32x8 acc[4];
#pragma unroll
  for (int i = 0; i < 8; ++i) { mrow[i] = -1e30f; lrow[i] = 0.0f; }
#pragma unroll
  for (int j = 0; j < 4; ++j) acc[j] = vzero8();

  for (int kt = 0; kt < SS / 32; ++kt) {
    const int t0 = kt * 32;
    // ---- batch-load all K fragments, then 4 WMMAs for a 16x32 S-tile
    f16x16 bkf[2][2];
#pragma unroll
    for (int ct = 0; ct < 2; ++ct) {
      const _Float16* krow =
          kb + (size_t)(b * SS + t0 + 16 * ct + c) * DD + h * DK;
#pragma unroll
      for (int dc = 0; dc < 2; ++dc) {
        const f16x8 lo = *(const f16x8*)(krow + 32 * dc + 16 * g);
        const f16x8 hi = *(const f16x8*)(krow + 32 * dc + 16 * g + 8);
#pragma unroll
        for (int e = 0; e < 8; ++e) { bkf[ct][dc][e] = lo[e]; bkf[ct][dc][e + 8] = hi[e]; }
      }
    }
    f32x8 sc[2];
#pragma unroll
    for (int ct = 0; ct < 2; ++ct) {
      f32x8 z = vzero8();
      z = __builtin_amdgcn_wmma_f32_16x16x32_f16(false, aq[0], false, bkf[ct][0],
                                                 (short)0, z, false, false);
      z = __builtin_amdgcn_wmma_f32_16x16x32_f16(false, aq[1], false, bkf[ct][1],
                                                 (short)0, z, false, false);
#pragma unroll
      for (int i = 0; i < 8; ++i) z[i] *= 0.03125f;  // 1/((D/H)/2)
      sc[ct] = z;
    }
    // ---- online softmax update (cross-lane within 16-lane groups)
    float alpha[8];
#pragma unroll
    for (int i = 0; i < 8; ++i) {
      float tm = fmaxf(sc[0][i], sc[1][i]);
#pragma unroll
      for (int off = 1; off < 16; off <<= 1)
        tm = fmaxf(tm, __shfl_xor(tm, off, 32));
      float mn = fmaxf(mrow[i], tm);
      alpha[i] = __expf(mrow[i] - mn);
      float p0 = __expf(sc[0][i] - mn);
      float p1 = __expf(sc[1][i] - mn);
      float rs = p0 + p1;
#pragma unroll
      for (int off = 1; off < 16; off <<= 1)
        rs += __shfl_xor(rs, off, 32);
      lrow[i] = lrow[i] * alpha[i] + rs;
      mrow[i] = mn;
      const int row = i + 8 * g;
      pw[row * 40 + c]      = (_Float16)p0;  // k = c
      pw[row * 40 + 16 + c] = (_Float16)p1;  // k = 16 + c
    }
#pragma unroll
    for (int j = 0; j < 4; ++j)
#pragma unroll
      for (int i = 0; i < 8; ++i) acc[j][i] *= alpha[i];

    // ---- P @ V (A from per-wave LDS bounce, B from VT: contiguous b128)
    f16x16 ap;
    {
      const f16x8 lo = *(const f16x8*)(pw + c * 40 + 16 * g);
      const f16x8 hi = *(const f16x8*)(pw + c * 40 + 16 * g + 8);
#pragma unroll
      for (int e = 0; e < 8; ++e) { ap[e] = lo[e]; ap[e + 8] = hi[e]; }
    }
    f16x16 bvf[4];
#pragma unroll
    for (int j = 0; j < 4; ++j) {
      const _Float16* vrow =
          vt + ((size_t)((b * HH + h) * DK + 16 * j + c)) * SS + t0 + 16 * g;
      const f16x8 lo = *(const f16x8*)(vrow);
      const f16x8 hi = *(const f16x8*)(vrow + 8);
#pragma unroll
      for (int e = 0; e < 8; ++e) { bvf[j][e] = lo[e]; bvf[j][e + 8] = hi[e]; }
    }
#pragma unroll
    for (int j = 0; j < 4; ++j)
      acc[j] = __builtin_amdgcn_wmma_f32_16x16x32_f16(false, ap, false, bvf[j],
                                                      (short)0, acc[j], false, false);
  }

  // ---- epilogue: divide by row sums, write f16 ctx [B,S,D]
  float linv[8];
#pragma unroll
  for (int i = 0; i < 8; ++i) linv[i] = 1.0f / lrow[i];
#pragma unroll
  for (int j = 0; j < 4; ++j)
#pragma unroll
    for (int i = 0; i < 8; ++i) {
      const int row = q0 + 8 * g + i;
      ctx[(size_t)(b * SS + row) * DD + h * DK + 16 * j + c] =
          (_Float16)(acc[j][i] * linv[i]);
    }
}

// =====================================================================
// Global LayerNorm over ALL 4M elements (deterministic 3-stage)
// =====================================================================
__global__ __launch_bounds__(256) void add_reduce_kernel(
    const float* __restrict__ a, const float* __restrict__ b,
    float* __restrict__ y, float* __restrict__ psum, float* __restrict__ psq) {
  __shared__ float ssum[256], ssq[256];
  const int tid = threadIdx.x;
  const size_t base = (size_t)blockIdx.x * 1024;
  float s = 0.0f, q = 0.0f;
#pragma unroll
  for (int i = 0; i < 4; ++i) {
    size_t idx = base + (size_t)i * 256 + tid;
    float v = a[idx] + b[idx];
    y[idx] = v;
    s += v; q += v * v;
  }
  ssum[tid] = s; ssq[tid] = q;
  __syncthreads();
  for (int off = 128; off > 0; off >>= 1) {
    if (tid < off) { ssum[tid] += ssum[tid + off]; ssq[tid] += ssq[tid + off]; }
    __syncthreads();
  }
  if (tid == 0) { psum[blockIdx.x] = ssum[0]; psq[blockIdx.x] = ssq[0]; }
}

__global__ __launch_bounds__(256) void ln_finalize_kernel(
    const float* __restrict__ psum, const float* __restrict__ psq,
    float* __restrict__ stats) {
  __shared__ float ssum[256], ssq[256];
  const int tid = threadIdx.x;
  float s = 0.0f, q = 0.0f;
#pragma unroll
  for (int i = 0; i < LN_BLOCKS / 256; ++i) {
    s += psum[i * 256 + tid];
    q += psq[i * 256 + tid];
  }
  ssum[tid] = s; ssq[tid] = q;
  __syncthreads();
  for (int off = 128; off > 0; off >>= 1) {
    if (tid < off) { ssum[tid] += ssum[tid + off]; ssq[tid] += ssq[tid + off]; }
    __syncthreads();
  }
  if (tid == 0) {
    const float invN = 1.0f / (float)((size_t)BB * SS * DD);
    float mu = ssum[0] * invN;
    float var = ssq[0] * invN - mu * mu;
    stats[0] = mu;
    stats[1] = rsqrtf(var + EPS);
  }
}

__global__ __launch_bounds__(256) void normalize_kernel(
    const float* __restrict__ y, const float* __restrict__ stats,
    float* __restrict__ outF, _Float16* __restrict__ outH) {
  const float mu = stats[0], r = stats[1];
  const size_t base = (size_t)blockIdx.x * 1024;
#pragma unroll
  for (int i = 0; i < 4; ++i) {
    size_t idx = base + (size_t)i * 256 + threadIdx.x;
    float v = (y[idx] - mu) * r;
    outF[idx] = v;
    if (outH) outH[idx] = (_Float16)v;
  }
}

// =====================================================================
// Host-side launch sequence
// =====================================================================
extern "C" void kernel_launch(void* const* d_in, const int* in_sizes, int n_in,
                              void* d_out, int out_size, void* d_ws, size_t ws_size,
                              hipStream_t stream) {
  const float* data = (const float*)d_in[0];
  const float* Wq = (const float*)d_in[1];
  const float* bq = (const float*)d_in[2];
  const float* Wk = (const float*)d_in[3];
  const float* bk = (const float*)d_in[4];
  const float* Wv = (const float*)d_in[5];
  const float* bv = (const float*)d_in[6];
  const float* Wo = (const float*)d_in[7];
  const float* bo = (const float*)d_in[8];
  const float* W1 = (const float*)d_in[9];
  const float* b1 = (const float*)d_in[10];
  const float* W2 = (const float*)d_in[11];
  const float* b2 = (const float*)d_in[12];

  const size_t MB = 1u << 20;
  const size_t NE = (size_t)BB * SS * DD;  // 4,194,304
  const int M = BB * SS;                   // 8192

  char* base = (char*)d_ws;
  // region 0..32MB: QH/KH/VT/CTXH (f16), later reused by FFH (f16, 32MB)
  _Float16* QH   = (_Float16*)(base + 0 * MB);
  _Float16* KH   = (_Float16*)(base + 8 * MB);
  _Float16* VT   = (_Float16*)(base + 16 * MB);
  _Float16* CTXH = (_Float16*)(base + 24 * MB);
  _Float16* FFH  = (_Float16*)(base + 0 * MB);   // reuse (QKV/CTX dead)
  float* MHA = (float*)(base + 32 * MB);          // 16MB
  float* FF2 = MHA;                               // reuse (MHA dead)
  float* Y1  = (float*)(base + 48 * MB);          // 16MB
  float* Y2  = Y1;                                // reuse (Y1 dead)
  float* X   = (float*)(base + 64 * MB);          // 16MB
  _Float16* DH = (_Float16*)(base + 80 * MB);     // 8MB
  _Float16* XH = DH;                              // reuse (DH dead)
  _Float16* WqT = (_Float16*)(base + 88 * MB);    // 512KB each
  _Float16* WkT = WqT + (size_t)DD * DD;
  _Float16* WvT = WkT + (size_t)DD * DD;
  _Float16* WoT = WvT + (size_t)DD * DD;
  _Float16* W1T = WoT + (size_t)DD * DD;          // [FF][D] = 2MB
  _Float16* W2T = W1T + (size_t)DD * FFD;         // [D][FF] = 2MB
  float* psum  = (float*)(base + 94 * MB);
  float* psq   = psum + LN_BLOCKS;
  float* stats = psq + LN_BLOCKS;

  dim3 blk(256);
  dim3 gD(DD / 128, M / 64);   // 4 x 128 : [8192,512]
  dim3 gF(FFD / 128, M / 64);  // 16 x 128: [8192,2048]

  // --- precision prep: activations + weights to f16 (weights transposed)
  cvt_f16_kernel<<<NE / 1024, blk, 0, stream>>>(data, DH);
  wtrans_kernel<<<dim3(DD / 32, DD / 32), blk, 0, stream>>>(Wq, WqT, DD, DD);
  wtrans_kernel<<<dim3(DD / 32, DD / 32), blk, 0, stream>>>(Wk, WkT, DD, DD);
  wtrans_kernel<<<dim3(DD / 32, DD / 32), blk, 0, stream>>>(Wv, WvT, DD, DD);
  wtrans_kernel<<<dim3(DD / 32, DD / 32), blk, 0, stream>>>(Wo, WoT, DD, DD);
  wtrans_kernel<<<dim3(FFD / 32, DD / 32), blk, 0, stream>>>(W1, W1T, DD, FFD);
  wtrans_kernel<<<dim3(DD / 32, FFD / 32), blk, 0, stream>>>(W2, W2T, FFD, DD);

  // --- QKV projections (f16 out; V written head-transposed)
  gemm_f16_kernel<<<gD, blk, 0, stream>>>(DH, WqT, bq, nullptr, QH, nullptr, M, DD, DD, 0);
  gemm_f16_kernel<<<gD, blk, 0, stream>>>(DH, WkT, bk, nullptr, KH, nullptr, M, DD, DD, 0);
  gemm_f16_kernel<<<gD, blk, 0, stream>>>(DH, WvT, bv, nullptr, nullptr, VT, M, DD, DD, 0);

  // --- attention
  flash_attn_kernel<<<dim3(SS / 64, HH, BB), dim3(128), 0, stream>>>(QH, KH, VT, CTXH);

  // --- output projection (f32 for residual)
  gemm_f16_kernel<<<gD, blk, 0, stream>>>(CTXH, WoT, bo, MHA, nullptr, nullptr, M, DD, DD, 0);

  // --- LN1 (global stats): x = LN(mha + data); emit f32 + f16
  add_reduce_kernel<<<LN_BLOCKS, blk, 0, stream>>>(MHA, data, Y1, psum, psq);
  ln_finalize_kernel<<<1, blk, 0, stream>>>(psum, psq, stats);
  normalize_kernel<<<LN_BLOCKS, blk, 0, stream>>>(Y1, stats, X, XH);

  // --- FFN
  gemm_f16_kernel<<<gF, blk, 0, stream>>>(XH, W1T, b1, nullptr, FFH, nullptr, M, FFD, DD, 1);
  gemm_f16_kernel<<<gD, blk, 0, stream>>>(FFH, W2T, b2, FF2, nullptr, nullptr, M, DD, FFD, 0);

  // --- LN2 -> final output
  add_reduce_kernel<<<LN_BLOCKS, blk, 0, stream>>>(X, FF2, Y2, psum, psq);
  ln_finalize_kernel<<<1, blk, 0, stream>>>(psum, psq, stats);
  normalize_kernel<<<LN_BLOCKS, blk, 0, stream>>>(Y2, stats, (float*)d_out, nullptr);
}
